// GATRegressor_6279242186855
// MI455X (gfx1250) — matrix operand order repaired and checked by reading, not compiled, not run on hardware
//
#include <hip/hip_runtime.h>

#define NNODES 100000
#define NEDGES 1600000
#define ETOT   (NEDGES + NNODES)   // reference concatenates self-loops
#define F      128                 // IN_C == HEADS*HID == 128
#define HEADS  4
#define HID    32
#define OUTC   2
#define NEG_SLOPE 0.2f

typedef _Float16 v16h __attribute__((ext_vector_type(16)));
typedef float    v8f  __attribute__((ext_vector_type(8)));

// ---------------------------------------------------------------------------
// WMMA GEMM: H[nrows x 128] = X[nrows x 128] * W[128 x 128]  (f32 in/out,
// f16 multiply, f32 accumulate). One block = 16-row strip, 8 waves = 8
// 16-col tiles, K loop of 4 x v_wmma_f32_16x16x32_f16.
// ---------------------------------------------------------------------------
__global__ __launch_bounds__(256) void k_gemm_wmma(
    const float* __restrict__ X, const float* __restrict__ W,
    float* __restrict__ H)
{
  __shared__ _Float16 sWt[F][F + 8];   // W transposed: sWt[n][k], padded
  __shared__ _Float16 sA [16][F + 8];  // A strip: sA[m][k]

  const int tid = threadIdx.x;
  const int rowBase = blockIdx.x * 16;

  // Stage W (f32 -> f16), transposed so B-fragment reads are k-contiguous.
  for (int i = tid; i < F * F; i += 256) {
    int k = i >> 7, n = i & 127;           // W row-major: W[k][n]
    sWt[n][k] = (_Float16)W[i];
  }
  // Stage the 16-row A strip.
  for (int i = tid; i < 16 * F; i += 256) {
    int m = i >> 7, k = i & 127;
    sA[m][k] = (_Float16)X[(long long)(rowBase + m) * F + k];
  }
  __syncthreads();

  const int lane = tid & 31;
  const int wid  = tid >> 5;
  const int colBase = wid * 16;
  const int half = lane >> 4;      // upper half-wave handles shifted K / M+8
  const int mrow = lane & 15;
  const int ncol = colBase + (lane & 15);

  v8f acc = {};
  #pragma unroll
  for (int ks = 0; ks < 4; ++ks) {
    v16h a, b;
    #pragma unroll
    for (int v = 0; v < 8; ++v) {
      // A 16x32 f16 layout: VGPR v<4 -> K base 0, v>=4 -> K base 16;
      // lanes 16-31 add +8.  (ISA 7.12.2)
      int ka = ks * 32 + ((v & 4) ? 16 : 0) + half * 8 + (v & 3) * 2;
      a[2 * v]     = sA[mrow][ka];
      a[2 * v + 1] = sA[mrow][ka + 1];
      // B 32x16 f16 layout: same K order as A (v<4 -> K base 0, v>=4 -> 16; lanes 16-31 +8).
      int kb = ks * 32 + ((v & 4) ? 16 : 0) + half * 8 + (v & 3) * 2;
      b[2 * v]     = sWt[ncol][kb];
      b[2 * v + 1] = sWt[ncol][kb + 1];
    }
    acc = __builtin_amdgcn_wmma_f32_16x16x32_f16(false, a, false, b,
                                                 (short)0, acc, false, false);
  }

  // C layout: VGPR r -> row M = r + 8*(lane>=16), col N = lane&15.
  #pragma unroll
  for (int r = 0; r < 8; ++r) {
    int row = rowBase + r + half * 8;
    H[(long long)row * F + ncol] = acc[r];
  }
}

// ---------------------------------------------------------------------------
// Per-(node,head) attention coefficients.
// ---------------------------------------------------------------------------
__global__ void k_alpha(const float* __restrict__ H,
                        const float* __restrict__ asrc,
                        const float* __restrict__ adst,
                        float* __restrict__ aS, float* __restrict__ aD)
{
  int i = blockIdx.x * blockDim.x + threadIdx.x;
  if (i >= NNODES * HEADS) return;
  int n = i >> 2, h = i & 3;
  const float* hp = &H[(long long)n * F + h * HID];
  float ss = 0.f, sd = 0.f;
  #pragma unroll
  for (int c = 0; c < HID; ++c) {
    float hv = hp[c];
    ss += hv * asrc[h * HID + c];
    sd += hv * adst[h * HID + c];
  }
  aS[i] = ss;
  aD[i] = sd;
}

// ---------------------------------------------------------------------------
// Init segment-softmax state + zero aggregation buffer.
// ---------------------------------------------------------------------------
__global__ void k_init(float* __restrict__ agg, float* __restrict__ mx,
                       float* __restrict__ dn)
{
  long long i = (long long)blockIdx.x * blockDim.x + threadIdx.x;
  if (i < (long long)NNODES * F) agg[i] = 0.f;
  if (i < (long long)NNODES * HEADS) { mx[i] = -3.0e38f; dn[i] = 0.f; }
}

__device__ inline void atomicMaxFloat(float* addr, float v)
{
  // Classic signed/unsigned split: correct total order over mixed signs.
  if (v >= 0.f) atomicMax((int*)addr, __float_as_int(v));
  else          atomicMin((unsigned int*)addr, __float_as_uint(v));
}

__device__ inline void edge_endpoints(const long long* __restrict__ ei,
                                      long long e, int& s, int& d)
{
  if (e < NEDGES) { s = (int)ei[e]; d = (int)ei[NEDGES + e]; }
  else            { s = d = (int)(e - NEDGES); }  // self-loop
}

// ---------------------------------------------------------------------------
// Segment max of leaky_relu(alpha_s[src]+alpha_d[dst]) over dst.
// ---------------------------------------------------------------------------
__global__ void k_edge_max(const long long* __restrict__ ei,
                           const float* __restrict__ aS,
                           const float* __restrict__ aD,
                           float* __restrict__ mx)
{
  long long e = (long long)blockIdx.x * blockDim.x + threadIdx.x;
  if (e >= ETOT) return;
  int s, d; edge_endpoints(ei, e, s, d);
  #pragma unroll
  for (int h = 0; h < HEADS; ++h) {
    float v = aS[s * HEADS + h] + aD[d * HEADS + h];
    v = v > 0.f ? v : v * NEG_SLOPE;
    atomicMaxFloat(&mx[d * HEADS + h], v);
  }
}

// ---------------------------------------------------------------------------
// Segment sum of exp(e - max) over dst.
// ---------------------------------------------------------------------------
__global__ void k_edge_sum(const long long* __restrict__ ei,
                           const float* __restrict__ aS,
                           const float* __restrict__ aD,
                           const float* __restrict__ mx,
                           float* __restrict__ dn)
{
  long long e = (long long)blockIdx.x * blockDim.x + threadIdx.x;
  if (e >= ETOT) return;
  int s, d; edge_endpoints(ei, e, s, d);
  #pragma unroll
  for (int h = 0; h < HEADS; ++h) {
    float v = aS[s * HEADS + h] + aD[d * HEADS + h];
    v = v > 0.f ? v : v * NEG_SLOPE;
    atomicAdd(&dn[d * HEADS + h], __expf(v - mx[d * HEADS + h]));
  }
}

// ---------------------------------------------------------------------------
// Weighted aggregation: agg[dst] += h[src] * alpha. One wave per edge;
// lane owns 4 channels (head = lane>>3). h/agg are L2-resident (2x51MB
// inside the 192MB L2), so gathers and f32 atomics stay on-chip.
// ---------------------------------------------------------------------------
__global__ __launch_bounds__(256) void k_edge_agg(
    const long long* __restrict__ ei,
    const float* __restrict__ aS, const float* __restrict__ aD,
    const float* __restrict__ mx, const float* __restrict__ dn,
    const float* __restrict__ H, float* __restrict__ agg)
{
  const int lane = threadIdx.x & 31;
  long long e = (long long)blockIdx.x * 8 + (threadIdx.x >> 5);
  if (e >= ETOT) return;
  int s, d; edge_endpoints(ei, e, s, d);

  const float* hrow = &H[(long long)s * F + lane * 4];
  __builtin_prefetch(hrow, 0, 0);  // global_prefetch_b8

  int h = lane >> 3;               // 4 channels/lane -> head = (lane*4)/32
  float v = aS[s * HEADS + h] + aD[d * HEADS + h];
  v = v > 0.f ? v : v * NEG_SLOPE;
  float alpha = __expf(v - mx[d * HEADS + h]) / (dn[d * HEADS + h] + 1e-16f);

  float4 hv = *(const float4*)hrow;
  float* op = &agg[(long long)d * F + lane * 4];
  atomicAdd(op + 0, hv.x * alpha);
  atomicAdd(op + 1, hv.y * alpha);
  atomicAdd(op + 2, hv.z * alpha);
  atomicAdd(op + 3, hv.w * alpha);
}

// ---------------------------------------------------------------------------
// out = elu(agg + bias)
// ---------------------------------------------------------------------------
__global__ void k_bias_elu(const float* __restrict__ agg,
                           const float* __restrict__ b,
                           float* __restrict__ dst)
{
  long long i = (long long)blockIdx.x * blockDim.x + threadIdx.x;
  if (i >= (long long)NNODES * F) return;
  float v = agg[i] + b[(int)(i & (F - 1))];
  dst[i] = v > 0.f ? v : (__expf(v) - 1.f);
}

// ---------------------------------------------------------------------------
// Final FC: [N,128] x [128,2] + bias.
// ---------------------------------------------------------------------------
__global__ void k_fc(const float* __restrict__ H,
                     const float* __restrict__ Wfc,
                     const float* __restrict__ bfc,
                     float* __restrict__ out)
{
  int n = blockIdx.x * blockDim.x + threadIdx.x;
  if (n >= NNODES) return;
  float a0 = bfc[0], a1 = bfc[1];
  const float* hp = &H[(long long)n * F];
  #pragma unroll 8
  for (int k = 0; k < F; ++k) {
    float hv = hp[k];
    a0 += hv * Wfc[k * 2 + 0];
    a1 += hv * Wfc[k * 2 + 1];
  }
  out[n * 2 + 0] = a0;
  out[n * 2 + 1] = a1;
}

// ---------------------------------------------------------------------------
extern "C" void kernel_launch(void* const* d_in, const int* in_sizes, int n_in,
                              void* d_out, int out_size, void* d_ws, size_t ws_size,
                              hipStream_t stream)
{
  (void)in_sizes; (void)n_in; (void)out_size; (void)ws_size;

  const float*     x   = (const float*)d_in[0];
  const long long* ei  = (const long long*)d_in[1];   // int64 [2,E]
  const float*     W1  = (const float*)d_in[2];
  const float*     as1 = (const float*)d_in[3];
  const float*     ad1 = (const float*)d_in[4];
  const float*     b1  = (const float*)d_in[5];
  const float*     W2  = (const float*)d_in[6];
  const float*     as2 = (const float*)d_in[7];
  const float*     ad2 = (const float*)d_in[8];
  const float*     b2  = (const float*)d_in[9];
  const float*     Wfc = (const float*)d_in[10];
  const float*     bfc = (const float*)d_in[11];
  float*           out = (float*)d_out;

  // Workspace layout (floats): two N*128 ping-pong buffers + 4 small arrays.
  float* bufA = (float*)d_ws;
  float* bufB = bufA + (size_t)NNODES * F;
  float* aS   = bufB + (size_t)NNODES * F;
  float* aD   = aS + (size_t)NNODES * HEADS;
  float* mx   = aD + (size_t)NNODES * HEADS;
  float* dn   = mx + (size_t)NNODES * HEADS;

  const int gGemm     = NNODES / 16;                 // 6250 (N % 16 == 0)
  const int gEdge     = (ETOT + 255) / 256;
  const int gEdgeAgg  = (ETOT + 7) / 8;              // 1 wave / edge
  const int gNodeHead = (NNODES * HEADS + 255) / 256;
  const int gNF       = (NNODES * F + 255) / 256;
  const int gNode     = (NNODES + 255) / 256;

  // ---------------- layer 1 ----------------
  k_gemm_wmma<<<gGemm, 256, 0, stream>>>(x, W1, bufA);
  k_alpha   <<<gNodeHead, 256, 0, stream>>>(bufA, as1, ad1, aS, aD);
  k_init    <<<gNF, 256, 0, stream>>>(bufB, mx, dn);
  k_edge_max<<<gEdge, 256, 0, stream>>>(ei, aS, aD, mx);
  k_edge_sum<<<gEdge, 256, 0, stream>>>(ei, aS, aD, mx, dn);
  k_edge_agg<<<gEdgeAgg, 256, 0, stream>>>(ei, aS, aD, mx, dn, bufA, bufB);
  k_bias_elu<<<gNF, 256, 0, stream>>>(bufB, b1, bufA);  // bufA = layer2 input

  // ---------------- layer 2 ----------------
  k_gemm_wmma<<<gGemm, 256, 0, stream>>>(bufA, W2, bufB);
  k_alpha   <<<gNodeHead, 256, 0, stream>>>(bufB, as2, ad2, aS, aD);
  k_init    <<<gNF, 256, 0, stream>>>(bufA, mx, dn);    // reuse bufA as agg
  k_edge_max<<<gEdge, 256, 0, stream>>>(ei, aS, aD, mx);
  k_edge_sum<<<gEdge, 256, 0, stream>>>(ei, aS, aD, mx, dn);
  k_edge_agg<<<gEdgeAgg, 256, 0, stream>>>(ei, aS, aD, mx, dn, bufB, bufA);
  k_bias_elu<<<gNF, 256, 0, stream>>>(bufA, b2, bufA);  // in-place elementwise

  // ---------------- FC ----------------
  k_fc<<<gNode, 256, 0, stream>>>(bufA, Wfc, bfc, out);
}
